// GNN_MD_23149873725632
// MI455X (gfx1250) — matrix operand order repaired and checked by reading, not compiled, not run on hardware
//
#include <hip/hip_runtime.h>
#include <hip/hip_bf16.h>

typedef __attribute__((ext_vector_type(16))) __bf16 v16bf;
typedef __attribute__((ext_vector_type(8)))  float  v8f;

#define GCN_N 50000
#define GCN_E 400000

static inline int ceil_div_i(long long a, long long b) { return (int)((a + b - 1) / b); }

// float -> bf16 bits, round to nearest even
__device__ __forceinline__ unsigned short f2bf_bits(float f) {
    union { float f; unsigned u; } a; a.f = f;
    unsigned u = a.u;
    unsigned r = u + 0x7FFFu + ((u >> 16) & 1u);
    return (unsigned short)(r >> 16);
}

// ---------------------------------------------------------------------------
// WMMA GEMM: D[N x F] = A[N x K] * W[K x F], A and W pre-converted to bf16.
//   Ab: [N x K] bf16 row-major (activations)
//   Wt: [F x K] bf16 row-major (weights transposed: lane's B fragment is one
//       contiguous 32B run, column n, K = kb2..kb2+15)
// One wave computes a 16x32 output tile (two WMMA tiles sharing the A frag).
// K is a template constant -> fully unrolled load+wmma stream.
// Epilogue modes (compile-time):
//   FUSE_AGG: write h AND  agg = selfnorm[row]*h + bias  (GCN layers; the
//             self-loop + bias init pass fused into the GEMM, saves a full
//             N*F read pass)
//   RELU    : write relu(h + bias)                        (FC1)
// ---------------------------------------------------------------------------
template<int K, bool FUSE_AGG, bool RELU>
__global__ void gemm_wmma(const unsigned short* __restrict__ Ab,
                          const unsigned short* __restrict__ Wt,
                          float* __restrict__ D,
                          float* __restrict__ Agg,
                          const float* __restrict__ selfn,
                          const float* __restrict__ bias,
                          int N, int F)
{
    const int tilesM = N >> 4;
    const int tilesN = F >> 5;             // 32-wide super-tiles
    const int wave = (int)((blockIdx.x * (unsigned)blockDim.x + threadIdx.x) >> 5);
    const int lane = threadIdx.x & 31;
    if (wave >= tilesM * tilesN) return;
    const int tm = wave % tilesM;          // adjacent waves share B tiles (L2 reuse)
    const int tn = wave / tilesM;
    const int half = lane >> 4;            // 0: lanes 0-15, 1: lanes 16-31
    const int l15  = lane & 15;
    const int m  = (tm << 4) + l15;        // A row owned by this lane
    const int n0 = (tn << 5) + l15;        // first B column owned by this lane
    const int n1 = n0 + 16;                // second B column

    v8f acc0 = {}, acc1 = {};
    union Frag { v16bf v; uint4 q[2]; };
    Frag a, b0, b1;

    // A: elems 0-15 -> K = kc + 16*half + (0..15), contiguous in Ab row.
    const unsigned short* ap  = Ab + (size_t)m  * K + (half << 4);
    // B: elems 0-15 -> K = kc + 16*half + (0..15), contiguous in Wt row.
    const unsigned short* bp0 = Wt + (size_t)n0 * K + (half << 4);
    const unsigned short* bp1 = Wt + (size_t)n1 * K + (half << 4);

    #pragma unroll
    for (int kc = 0; kc < K; kc += 32) {
        a.q[0]  = *(const uint4*)(ap  + kc);
        a.q[1]  = *(const uint4*)(ap  + kc + 8);
        b0.q[0] = *(const uint4*)(bp0 + kc);
        b0.q[1] = *(const uint4*)(bp0 + kc + 8);
        b1.q[0] = *(const uint4*)(bp1 + kc);
        b1.q[1] = *(const uint4*)(bp1 + kc + 8);
        acc0 = __builtin_amdgcn_wmma_f32_16x16x32_bf16(
                   false, a.v, false, b0.v, (short)0, acc0, false, false);
        acc1 = __builtin_amdgcn_wmma_f32_16x16x32_bf16(
                   false, a.v, false, b1.v, (short)0, acc1, false, false);
    }

    const float b0v = bias[n0];
    const float b1v = bias[n1];
    // C/D layout: VGPR r -> M = r (lanes 0-15) or r+8 (lanes 16-31), N = lane&15.
    #pragma unroll
    for (int r = 0; r < 8; ++r) {
        const int mm = (tm << 4) + r + (half << 3);
        float v0 = acc0[r];
        float v1 = acc1[r];
        if (FUSE_AGG) {
            D[(size_t)mm * F + n0] = v0;
            D[(size_t)mm * F + n1] = v1;
            const float sn = selfn[mm];
            Agg[(size_t)mm * F + n0] = sn * v0 + b0v;
            Agg[(size_t)mm * F + n1] = sn * v1 + b1v;
        } else {
            v0 += b0v; v1 += b1v;
            if (RELU) { v0 = fmaxf(v0, 0.0f); v1 = fmaxf(v1, 0.0f); }
            D[(size_t)mm * F + n0] = v0;
            D[(size_t)mm * F + n1] = v1;
        }
    }
}

// ---------------------------------------------------------------------------
// Conversions
// ---------------------------------------------------------------------------
__global__ void f32_to_bf16(const float* __restrict__ src,
                            unsigned short* __restrict__ dst, long long n) {
    long long t = (long long)blockIdx.x * blockDim.x + threadIdx.x;
    if (t < n) dst[t] = f2bf_bits(src[t]);
}

// W [K x F] f32 -> Wt [F x K] bf16 (transpose + convert, once per launch)
__global__ void w_transpose_bf16(const float* __restrict__ W,
                                 unsigned short* __restrict__ Wt, int K, int F) {
    int t = blockIdx.x * blockDim.x + threadIdx.x;
    if (t >= K * F) return;
    const int k = t / F, f = t % F;
    Wt[(size_t)f * K + k] = f2bf_bits(W[t]);
}

// ---------------------------------------------------------------------------
// Degree / normalization precompute
// ---------------------------------------------------------------------------
__global__ void deg_init(float* __restrict__ deg, int N) {
    int i = blockIdx.x * blockDim.x + threadIdx.x;
    if (i < N) deg[i] = 1.0f;              // self-loop weight
}
__global__ void deg_accum(float* __restrict__ deg, const int* __restrict__ col,
                          const float* __restrict__ w, int E) {
    int e = blockIdx.x * blockDim.x + threadIdx.x;
    if (e < E) atomicAdd(&deg[col[e]], w[e]);
}
__global__ void dinv_compute(float* __restrict__ deg, float* __restrict__ selfnorm, int N) {
    int i = blockIdx.x * blockDim.x + threadIdx.x;
    if (i >= N) return;
    float d = deg[i];
    float dv = (d > 0.0f) ? rsqrtf(d) : 0.0f;
    deg[i] = dv;                            // deg buffer now holds dinv
    selfnorm[i] = dv * dv;                  // self-loop norm = dinv^2 (weight 1)
}
__global__ void edge_norm(const float* __restrict__ dinv, const int* __restrict__ row,
                          const int* __restrict__ col, const float* __restrict__ w,
                          float* __restrict__ norm, int E) {
    int e = blockIdx.x * blockDim.x + threadIdx.x;
    if (e < E) norm[e] = dinv[row[e]] * w[e] * dinv[col[e]];
}

// ---------------------------------------------------------------------------
// Edge scatter: agg[col[e]] += norm[e] * h[row[e]]  (4 features per thread)
// h and agg live in the 192MB L2, so this is an L2-RMW-bound pass.
// ---------------------------------------------------------------------------
__global__ void scatter_edges(const float* __restrict__ h, const int* __restrict__ row,
                              const int* __restrict__ col, const float* __restrict__ norm,
                              float* __restrict__ agg, int E, int F) {
    const int chunks = F >> 2;
    long long t = (long long)blockIdx.x * blockDim.x + threadIdx.x;
    if (t >= (long long)E * chunks) return;
    const int e  = (int)(t / chunks);
    const int f0 = ((int)(t % chunks)) << 2;
    const float nv = norm[e];
    const float4 v = *(const float4*)(h + (size_t)row[e] * F + f0);
    float* ap = agg + (size_t)col[e] * F + f0;
    atomicAdd(ap + 0, nv * v.x);
    atomicAdd(ap + 1, nv * v.y);
    atomicAdd(ap + 2, nv * v.z);
    atomicAdd(ap + 3, nv * v.w);
}

// ---------------------------------------------------------------------------
// BatchNorm train-mode statistics, two-phase coalesced reduction.
// Phase 1: blocks sweep contiguous row chunks; lane -> feature mapping is
//          f = tid % F (coalesced; F divides blockDim). Partials in LDS,
//          flushed with global f32 atomics into sums[0..F)=Σx, sums[F..2F)=Σx².
// Phase 2: finalize mean / rstd.
// ---------------------------------------------------------------------------
__global__ void bn_zero(float* __restrict__ sums, int F) {
    int i = blockIdx.x * blockDim.x + threadIdx.x;
    if (i < 2 * F) sums[i] = 0.0f;
}

__global__ void bn_partial(const float* __restrict__ x, int N, int F,
                           float* __restrict__ sums, int rowsPerBlock, int prerelu) {
    extern __shared__ float sh[];          // 2*F floats
    for (int i = threadIdx.x; i < 2 * F; i += blockDim.x) sh[i] = 0.0f;
    __syncthreads();

    const int tpf  = blockDim.x / F;       // threads cooperating per feature
    const int f    = (int)threadIdx.x % F;
    const int rsub = (int)threadIdx.x / F;
    const long long r0 = (long long)blockIdx.x * rowsPerBlock;
    const long long r1 = (r0 + rowsPerBlock < N) ? (r0 + rowsPerBlock) : N;

    float s = 0.0f, q = 0.0f;
    for (long long r = r0 + rsub; r < r1; r += tpf) {
        float v = x[r * F + f];
        if (prerelu) v = fmaxf(v, 0.0f);
        s += v; q += v * v;
    }
    atomicAdd(&sh[f], s);
    atomicAdd(&sh[F + f], q);
    __syncthreads();
    for (int i = threadIdx.x; i < F; i += blockDim.x) {
        atomicAdd(&sums[i],     sh[i]);
        atomicAdd(&sums[F + i], sh[F + i]);
    }
}

__global__ void bn_finalize(const float* __restrict__ sums, int N, int F,
                            float* __restrict__ mean, float* __restrict__ rstd) {
    int f = blockIdx.x * blockDim.x + threadIdx.x;
    if (f >= F) return;
    const float mu  = sums[f] / (float)N;
    const float var = sums[F + f] / (float)N - mu * mu;   // biased variance
    mean[f] = mu;
    rstd[f] = rsqrtf(var + 1e-5f);
}

// BN normalize (+optional relu before/after), emitting bf16 for the next GEMM
__global__ void bn_apply_bf16(const float* __restrict__ x,
                              unsigned short* __restrict__ outb, int N, int F,
                              const float* __restrict__ mean, const float* __restrict__ rstd,
                              const float* __restrict__ g, const float* __restrict__ be,
                              int prerelu, int postrelu) {
    long long t = (long long)blockIdx.x * blockDim.x + threadIdx.x;
    if (t >= (long long)N * F) return;
    const int f = (int)(t % F);
    float v = x[t];
    if (prerelu) v = fmaxf(v, 0.0f);
    v = g[f] * (v - mean[f]) * rstd[f] + be[f];
    if (postrelu) v = fmaxf(v, 0.0f);
    outb[t] = f2bf_bits(v);
}

// ---------------------------------------------------------------------------
// Final FC2: out[i] = h[i,:] . w + b   (F_out = 1)
// ---------------------------------------------------------------------------
__global__ void fc2_dot(const float* __restrict__ h, const float* __restrict__ w,
                        const float* __restrict__ b, float* __restrict__ out,
                        int N, int K) {
    int i = blockIdx.x * blockDim.x + threadIdx.x;
    if (i >= N) return;
    float s = b[0];
    const float* hp = h + (size_t)i * K;
    #pragma unroll 4
    for (int k = 0; k < K; ++k) s += hp[k] * w[k];
    out[i] = s;
}

// ---------------------------------------------------------------------------
// Launch
// ---------------------------------------------------------------------------
extern "C" void kernel_launch(void* const* d_in, const int* in_sizes, int n_in,
                              void* d_out, int out_size, void* d_ws, size_t ws_size,
                              hipStream_t stream) {
    const int N = GCN_N, E = GCN_E;
    (void)n_in; (void)in_sizes; (void)ws_size; (void)out_size;

    const float* x  = (const float*)d_in[0];
    const int*   ei = (const int*)d_in[1];
    const float* ew = (const float*)d_in[2];
    const int* row = ei;           // edge_index[0]
    const int* col = ei + E;       // edge_index[1]

    // workspace partition
    float* buf_h   = (float*)d_ws;                        // N*256 f32 (GEMM out, h)
    float* buf_agg = buf_h + (size_t)N * 256;             // N*256 f32 (aggregate)
    float* norm    = buf_agg + (size_t)N * 256;           // E
    float* deg     = norm + E;                            // N  (becomes dinv)
    float* selfn   = deg + N;                             // N  (dinv^2)
    float* mean    = selfn + N;                           // 256
    float* rstd    = mean + 256;                          // 256
    float* bnsums  = rstd + 256;                          // 512 (Σx, Σx²)
    unsigned short* buf_bf  = (unsigned short*)(bnsums + 512); // N*256 bf16 (GEMM in)
    unsigned short* wt_pool = buf_bf + (size_t)N * 256;        // transposed bf16 weights

    const int B = 256;

    // ---- convert + transpose all weights to bf16 once ----
    struct WDesc { int K, F, wi; };
    const WDesc wd[6] = {
        {32, 32, 3}, {32, 64, 7}, {64, 128, 11},
        {128, 128, 15}, {128, 256, 19}, {256, 128, 23},
    };
    unsigned short* wt[6];
    {
        size_t off = 0;
        for (int i = 0; i < 6; ++i) {
            wt[i] = wt_pool + off;
            off += (size_t)wd[i].K * wd[i].F;
            w_transpose_bf16<<<ceil_div_i((long long)wd[i].K * wd[i].F, B), B, 0, stream>>>(
                (const float*)d_in[wd[i].wi], wt[i], wd[i].K, wd[i].F);
        }
    }

    // ---- degrees & symmetric normalization ----
    deg_init<<<ceil_div_i(N, B), B, 0, stream>>>(deg, N);
    deg_accum<<<ceil_div_i(E, B), B, 0, stream>>>(deg, col, ew, E);
    dinv_compute<<<ceil_div_i(N, B), B, 0, stream>>>(deg, selfn, N);
    edge_norm<<<ceil_div_i(E, B), B, 0, stream>>>(deg, row, col, ew, norm, E);

    // ---- layer-1 input to bf16 ----
    f32_to_bf16<<<ceil_div_i((long long)N * 32, B), B, 0, stream>>>(x, buf_bf, (long long)N * 32);

    // ---- 5 GCN layers: GEMM(+fused selfloop/bias agg-init) -> scatter -> BN ----
    const int prerelu_flags[5] = {1, 1, 1, 0, 0};   // BN(ReLU(.)) vs ReLU(BN(.))
    for (int l = 0; l < 5; ++l) {
        const int K = wd[l].K, F = wd[l].F, wi = wd[l].wi;
        const float* b  = (const float*)d_in[wi + 1];
        const float* g  = (const float*)d_in[wi + 2];
        const float* be = (const float*)d_in[wi + 3];

        const long long waves = (long long)(N >> 4) * (F >> 5);
        const int gblocks = ceil_div_i(waves * 32, B);
        switch (K) {   // h -> buf_h ; agg-init (selfnorm*h + bias) -> buf_agg
            case 32:  gemm_wmma<32,  true, false><<<gblocks, B, 0, stream>>>(
                          buf_bf, wt[l], buf_h, buf_agg, selfn, b, N, F); break;
            case 64:  gemm_wmma<64,  true, false><<<gblocks, B, 0, stream>>>(
                          buf_bf, wt[l], buf_h, buf_agg, selfn, b, N, F); break;
            case 128: gemm_wmma<128, true, false><<<gblocks, B, 0, stream>>>(
                          buf_bf, wt[l], buf_h, buf_agg, selfn, b, N, F); break;
        }

        scatter_edges<<<ceil_div_i((long long)E * (F >> 2), B), B, 0, stream>>>(
            buf_h, row, col, norm, buf_agg, E, F);

        const int pre = prerelu_flags[l], post = pre ? 0 : 1;
        const int rowsPerBlock = 128;
        bn_zero<<<ceil_div_i(2 * F, B), B, 0, stream>>>(bnsums, F);
        bn_partial<<<ceil_div_i(N, rowsPerBlock), B, 2 * F * sizeof(float), stream>>>(
            buf_agg, N, F, bnsums, rowsPerBlock, pre);
        bn_finalize<<<ceil_div_i(F, B), B, 0, stream>>>(bnsums, N, F, mean, rstd);
        bn_apply_bf16<<<ceil_div_i((long long)N * F, B), B, 0, stream>>>(
            buf_agg, buf_bf, N, F, mean, rstd, g, be, pre, post);
    }

    // ---- FC1 (256 -> 128), fused bias + relu ----
    {
        const long long waves = (long long)(N >> 4) * (128 >> 5);
        const int gblocks = ceil_div_i(waves * 32, B);
        gemm_wmma<256, false, true><<<gblocks, B, 0, stream>>>(
            buf_bf, wt[5], buf_h, nullptr, nullptr, (const float*)d_in[24], N, 128);
    }

    // ---- FC2 (128 -> 1) ----
    fc2_dot<<<ceil_div_i(N, B), B, 0, stream>>>(
        buf_h, (const float*)d_in[25], (const float*)d_in[26], (float*)d_out, N, 128);
}
